// IntSoftmax_49615462204005
// MI455X (gfx1250) — compile-verified, matrix operand order checked by reading, not run on hardware
//
#include <hip/hip_runtime.h>
#include <stdint.h>

// ---------------------------------------------------------------------------
// Integer softmax, one 64-thread block (2 wave32) per row of 2048 elements.
// HBM-bound (~1 GiB traffic): single pass, b128 NT loads/stores.
// Row-sum of 8-bit exp values done with v_wmma_i32_16x16x64_iu8 (B = ones).
// Exp LUT staged to LDS via global_load_async_to_lds_b128 (builtin confirmed
// present by round-1 diagnostic; signature is (v4i AS1*, v4i AS3*, Ii, Ii)).
// ---------------------------------------------------------------------------

typedef __attribute__((ext_vector_type(8))) int   v8i;
typedef __attribute__((ext_vector_type(4))) int   i4v;
typedef __attribute__((ext_vector_type(4))) float f4v;

// Types matching the async-LDS builtin's parameters (gcc vector_size flavor).
typedef int v4i_vs __attribute__((vector_size(16)));
typedef __attribute__((address_space(1))) v4i_vs* gptr_v4i;
typedef __attribute__((address_space(3))) v4i_vs* lptr_v4i;

#define ROW_LEN   2048
#define TPB       64
#define CHUNKS    8      // 8 x int4 per thread = 32 elements
#define TRUNC_MUL 12

#if defined(__has_builtin)
#if __has_builtin(__builtin_amdgcn_global_load_async_to_lds_b128)
#define HAVE_ASYNC_LDS 1
#endif
#endif

__device__ __forceinline__ int imax(int a, int b) { return a > b ? a : b; }
__device__ __forceinline__ int imin(int a, int b) { return a < b ? a : b; }

__global__ __launch_bounds__(TPB)
void int_softmax_rows(const int* __restrict__ t,
                      const int* __restrict__ mask,
                      const int* __restrict__ exp_table,
                      const int* __restrict__ exp_shift_p,
                      const int* __restrict__ recip_tables,
                      const int* __restrict__ recip_alphas,
                      const int* __restrict__ recip_betas,
                      const int* __restrict__ recip_log2denoms,
                      const int* __restrict__ recip_offsets_diff,
                      float* __restrict__ out,
                      int rows_per_batch)
{
    __shared__ int s_exp[256];   // staged exp LUT
    __shared__ int s_red[4];     // [0..1] per-wave max, [2..3] per-wave sum
    __shared__ int s_r;          // broadcast reciprocal

    const int     tid  = threadIdx.x;
    const int     lane = tid & 31;
    const int     wave = tid >> 5;
    const int64_t row  = blockIdx.x;
    const int     b    = (int)(row / (int64_t)rows_per_batch);

    // ---- stage exp table (256 ints = 1 KB) into LDS: 64 lanes x b128 ------
#if HAVE_ASYNC_LDS
    {
        int* src_nc = const_cast<int*>(exp_table + tid * 4);
        __builtin_amdgcn_global_load_async_to_lds_b128(
            (gptr_v4i)src_nc,
            (lptr_v4i)(&s_exp[tid * 4]),
            /*offset=*/0, /*cpol=*/0);
#if __has_builtin(__builtin_amdgcn_s_wait_asynccnt)
        __builtin_amdgcn_s_wait_asynccnt(0);
#else
        asm volatile("s_wait_asynccnt 0" ::: "memory");
#endif
    }
#else
    ((i4v*)s_exp)[tid] = ((const i4v*)exp_table)[tid];
#endif

    const int exp_shift = exp_shift_p[0];

    // ---- load 32 elements/thread, coalesced b128, add mask, local max -----
    const int* trow = t    + row * (int64_t)ROW_LEN;
    const int* mrow = mask + (int64_t)b * ROW_LEN;

    i4v tv[CHUNKS];
    int tmax = INT32_MIN;
#pragma unroll
    for (int it = 0; it < CHUNKS; ++it) {
        const int base = it * (TPB * 4) + tid * 4;   // lane-contiguous 16B
        i4v v = __builtin_nontemporal_load((const i4v*)(trow + base));
        i4v m = *(const i4v*)(mrow + base);          // temporal: reused, L2-resident
        v.x += m.x; v.y += m.y; v.z += m.z; v.w += m.w;
        tv[it] = v;
        tmax = imax(tmax, imax(imax(v.x, v.y), imax(v.z, v.w)));
    }

    __syncthreads();                                  // s_exp ready

    // ---- block-wide max: wave32 xor-shuffle tree + 2-wave LDS exchange ----
#pragma unroll
    for (int off = 16; off > 0; off >>= 1)
        tmax = imax(tmax, __shfl_xor(tmax, off, 32));
    if (lane == 0) s_red[wave] = tmax;
    __syncthreads();
    tmax = imax(s_red[0], s_red[1]);

    // ---- exp LUT gather from LDS; pack e (<=255) into IU8 A-matrix --------
    v8i amat;
#pragma unroll
    for (int it = 0; it < CHUNKS; ++it) {
        const i4v v  = tv[it];
        const int e0 = s_exp[imin((tmax - v.x) >> exp_shift, 255)];
        const int e1 = s_exp[imin((tmax - v.y) >> exp_shift, 255)];
        const int e2 = s_exp[imin((tmax - v.z) >> exp_shift, 255)];
        const int e3 = s_exp[imin((tmax - v.w) >> exp_shift, 255)];
        amat[it] = (e0 & 255) | ((e1 & 255) << 8) | ((e2 & 255) << 16) | ((e3 & 255) << 24);
    }

    // ---- row sum via WMMA: D = A(16x64 u8) x ones(64x16 u8) ---------------
    // Each wave holds exactly 1024 e-bytes = one full A tile; B = all-ones so
    // D[m][n] = sum_k A[m][k]. Total = sum of D over rows = (lane0 + lane16)
    // partials of the 8 D VGPRs.
    v8i bmat;
#pragma unroll
    for (int i = 0; i < 8; ++i) bmat[i] = 0x01010101;
    v8i cacc = {};
    cacc = __builtin_amdgcn_wmma_i32_16x16x64_iu8(
        /*sgn_a=*/false, amat, /*sgn_b=*/false, bmat, cacc,
        /*reuse_a=*/false, /*reuse_b=*/false);

    int part = cacc[0] + cacc[1] + cacc[2] + cacc[3]
             + cacc[4] + cacc[5] + cacc[6] + cacc[7];
    const int wave_sum = __shfl(part, 0, 32) + __shfl(part, 16, 32);
    if (lane == 0) s_red[2 + wave] = wave_sum;
    __syncthreads();
    const int s = s_red[2] + s_red[3];

    // ---- segmented reciprocal LUT (thread 0), broadcast r -----------------
    if (tid == 0) {
        int lut = 0;
#pragma unroll
        for (int i = 0; i < 4; ++i)
            if (s >= recip_alphas[i] && s < recip_betas[i]) lut = i;
        int ridx = (s - recip_alphas[lut]) >> recip_log2denoms[lut];
        ridx = imin(imax(ridx, 0), 255);
        s_r = recip_tables[lut * 256 + ridx] << recip_offsets_diff[lut];
    }
    __syncthreads();
    const int r = s_r;

    // ---- out = (e * r) >> 12, non-temporal b128 stores --------------------
    float* orow = out + row * (int64_t)ROW_LEN;
#pragma unroll
    for (int it = 0; it < CHUNKS; ++it) {
        const int base = it * (TPB * 4) + tid * 4;
        const int p    = amat[it];
        f4v o;
        o.x = (float)(((( p        ) & 255) * r) >> TRUNC_MUL);
        o.y = (float)((((p >>  8) & 255) * r) >> TRUNC_MUL);
        o.z = (float)((((p >> 16) & 255) * r) >> TRUNC_MUL);
        o.w = (float)((((p >> 24) & 255) * r) >> TRUNC_MUL);
        __builtin_nontemporal_store(o, (f4v*)(orow + base));
    }
}

extern "C" void kernel_launch(void* const* d_in, const int* in_sizes, int n_in,
                              void* d_out, int out_size, void* d_ws, size_t ws_size,
                              hipStream_t stream)
{
    const int* t    = (const int*)d_in[0];
    const int* mask = (const int*)d_in[1];
    const int* expt = (const int*)d_in[2];
    const int* exps = (const int*)d_in[3];
    const int* rtab = (const int*)d_in[4];
    const int* ralf = (const int*)d_in[5];
    const int* rbet = (const int*)d_in[6];
    const int* rld  = (const int*)d_in[7];
    const int* roff = (const int*)d_in[8];
    float*     out  = (float*)d_out;

    const int total_rows     = in_sizes[0] / ROW_LEN;  // B*H*S = 65536
    const int batches        = in_sizes[1] / ROW_LEN;  // B = 2
    const int rows_per_batch = total_rows / batches;   // H*S

    hipLaunchKernelGGL(int_softmax_rows, dim3(total_rows), dim3(TPB), 0, stream,
                       t, mask, expt, exps, rtab, ralf, rbet, rld, roff, out,
                       rows_per_batch);
}
